// VAE_Autoencoder_59176059404336
// MI455X (gfx1250) — matrix operand order repaired and checked by reading, not compiled, not run on hardware
//
#include <hip/hip_runtime.h>
#include <hip/hip_bf16.h>

// ---------------------------------------------------------------------------
// GRU-VAE forward for MI455X (gfx1250, wave32, WMMA + async global->LDS DMA).
// Dims: V=E=L=256, H=1024, B=1024, S=MAX_LEN=256.
// ---------------------------------------------------------------------------

typedef __bf16 bf16;
typedef __attribute__((ext_vector_type(16))) __bf16 v16bf;
typedef __attribute__((ext_vector_type(8)))  float  v8f;

#define BQ   1024   // batch
#define HQ   1024   // hidden
#define EQ   256    // embed
#define VQ   256    // vocab
#define LQ   256    // latent
#define SQ   256    // seq len
#define MAXL 256    // decode len

// GEMM block tile
#define TM 64
#define TN 256
#define TK 32

// ----------------------- CDNA5 async copy primitives -----------------------
// global_load_async_to_lds_b128: memory -> LDS DMA, tracked by ASYNCcnt.
// VDST VGPR holds the workgroup-relative LDS byte address (= low 32 bits of
// the generic pointer: flat LDS aperture keeps the LDS offset in addr[31:0]).
__device__ __forceinline__ void async_ld_b128(void* lds_dst, const void* gsrc) {
  uint32_t ldsoff = (uint32_t)(uintptr_t)lds_dst;
  uint64_t gaddr  = (uint64_t)(uintptr_t)gsrc;
  asm volatile("global_load_async_to_lds_b128 %0, %1, off"
               :: "v"(ldsoff), "v"(gaddr)
               : "memory");
}

template <int N>
__device__ __forceinline__ void wait_asynccnt() {
  asm volatile("s_wait_asynccnt %0" :: "i"(N) : "memory");
}

// ------------------------- WMMA fragment loaders ---------------------------
// A fragment: 16x32 bf16 (M x K).  ISA 7.12.2: lanes 0-15 -> M=lane, K groups
// {0-7,16-23}; lanes 16-31 -> same M, K groups {8-15,24-31}; 2 K per VGPR.
__device__ __forceinline__ v16bf load_frag_A(const bf16* __restrict__ As,
                                             int rowBase, int lane) {
  const int m = rowBase + (lane & 15);
  const int khalf = (lane >> 4) << 3;          // 0 or 8
  v16bf a;
#pragma unroll
  for (int j = 0; j < 16; ++j) {
    const int vg = j >> 1, lo = j & 1;
    const int k = ((vg & 4) ? 16 : 0) + khalf + ((vg & 3) << 1) + lo;
    a[j] = As[m * TK + k];
  }
  return a;
}

// B fragment: 32x16 bf16 (K x N).  Lanes 0-15 -> N=lane, K groups {0-7,16-23};
// lanes 16-31 -> same N, K groups {8-15,24-31} (matches the A fragment).
__device__ __forceinline__ v16bf load_frag_B(const bf16* __restrict__ Bs,
                                             int colBase, int lane) {
  const int n = colBase + (lane & 15);
  const int kbase = (lane >> 4) << 3;          // 0 or 8
  v16bf b;
#pragma unroll
  for (int j = 0; j < 16; ++j) {
    b[j] = Bs[n * TK + ((j & 8) ? 16 : 0) + kbase + (j & 7)];
  }
  return b;
}

// ------------------------------- GEMM --------------------------------------
// C[M,N](f32, row stride ldc) = A[M,K](bf16, row stride lda) * W[N,K]^T + bias
// Block tile 64x256, 8 wave32 waves (2x4); wave tile 32x64 = 2x4 WMMA tiles.
// Global->LDS staging uses async DMA (ASYNCcnt), double-buffered.
// Requires M%64==0, N%256==0, K%32==0 (true for all call sites here).
__global__ __launch_bounds__(256) void gemm_bf16_wmma(
    const bf16* __restrict__ A, int lda,
    const bf16* __restrict__ W, int ldw,
    const float* __restrict__ bias,
    float* __restrict__ C, int ldc, int K) {
  __shared__ __align__(16) bf16 As[2][TM * TK];   //  2 x  4 KB
  __shared__ __align__(16) bf16 Bs[2][TN * TK];   //  2 x 16 KB

  const int tid  = threadIdx.x;
  const int lane = tid & 31;
  const int wave = tid >> 5;             // 0..7
  const int wm   = (wave >> 2) * 32;     // 0 / 32
  const int wn   = (wave & 3) * 64;      // 0 / 64 / 128 / 192
  const int m0   = blockIdx.y * TM;
  const int n0   = blockIdx.x * TN;

  // stage issue: 1 A-chunk + 4 B-chunks of b128 per thread (5 async ops)
  auto stage = [&](int buf, int k0) {
    {
      const int idx = tid * 8;                   // 0..2047
      const int r = idx >> 5, c = idx & 31;
      async_ld_b128(&As[buf][idx], &A[(size_t)(m0 + r) * lda + k0 + c]);
    }
#pragma unroll
    for (int i = 0; i < 4; ++i) {
      const int idx = (tid + i * 256) * 8;       // 0..8191
      const int r = idx >> 5, c = idx & 31;
      async_ld_b128(&Bs[buf][idx], &W[(size_t)(n0 + r) * ldw + k0 + c]);
    }
  };

  v8f acc[2][4] = {};

  stage(0, 0);                                   // prologue: tile 0 in flight

  for (int k0 = 0; k0 < K; k0 += TK) {
    const int buf  = (k0 >> 5) & 1;
    const bool more = (k0 + TK) < K;
    if (more) stage(buf ^ 1, k0 + TK);           // overlap next tile DMA

    // keep L2 warm two tiles ahead (speculative, no counters)
    if ((k0 + 2 * TK) < K) {
      __builtin_prefetch(&W[(size_t)(n0 + tid) * ldw + k0 + 2 * TK], 0, 1);
    }

    if (more) wait_asynccnt<5>();                // 5 newer copies in flight
    else      wait_asynccnt<0>();
    __syncthreads();                             // tile `buf` visible to all

    v16bf a0 = load_frag_A(As[buf], wm,      lane);
    v16bf a1 = load_frag_A(As[buf], wm + 16, lane);
    v16bf bf0 = load_frag_B(Bs[buf], wn,      lane);
    v16bf bf1 = load_frag_B(Bs[buf], wn + 16, lane);
    v16bf bf2 = load_frag_B(Bs[buf], wn + 32, lane);
    v16bf bf3 = load_frag_B(Bs[buf], wn + 48, lane);

    acc[0][0] = __builtin_amdgcn_wmma_f32_16x16x32_bf16(
        false, a0, false, bf0, (short)0, acc[0][0], false, false);
    acc[0][1] = __builtin_amdgcn_wmma_f32_16x16x32_bf16(
        false, a0, false, bf1, (short)0, acc[0][1], false, false);
    acc[0][2] = __builtin_amdgcn_wmma_f32_16x16x32_bf16(
        false, a0, false, bf2, (short)0, acc[0][2], false, false);
    acc[0][3] = __builtin_amdgcn_wmma_f32_16x16x32_bf16(
        false, a0, false, bf3, (short)0, acc[0][3], false, false);
    acc[1][0] = __builtin_amdgcn_wmma_f32_16x16x32_bf16(
        false, a1, false, bf0, (short)0, acc[1][0], false, false);
    acc[1][1] = __builtin_amdgcn_wmma_f32_16x16x32_bf16(
        false, a1, false, bf1, (short)0, acc[1][1], false, false);
    acc[1][2] = __builtin_amdgcn_wmma_f32_16x16x32_bf16(
        false, a1, false, bf2, (short)0, acc[1][2], false, false);
    acc[1][3] = __builtin_amdgcn_wmma_f32_16x16x32_bf16(
        false, a1, false, bf3, (short)0, acc[1][3], false, false);

    __syncthreads();                             // frag reads done before next
  }                                              // stage overwrites the buffer

  // --- epilogue: C layout, lanes 0-15 M=r, lanes 16-31 M=8+r; N=lane&15 ---
#pragma unroll
  for (int i = 0; i < 2; ++i) {
#pragma unroll
    for (int j = 0; j < 4; ++j) {
#pragma unroll
      for (int r = 0; r < 8; ++r) {
        const int row = m0 + wm + i * 16 + (lane >> 4) * 8 + r;
        const int col = n0 + wn + j * 16 + (lane & 15);
        float v = acc[i][j][r];
        if (bias) v += bias[col];
        C[(size_t)row * ldc + col] = v;
      }
    }
  }
}

// --------------------------- pointwise kernels -----------------------------
__global__ void cast_f32_bf16(const float* __restrict__ in,
                              bf16* __restrict__ out, int n) {
  int i = blockIdx.x * blockDim.x + threadIdx.x;
  if (i < n) out[i] = (bf16)in[i];
}

__global__ void zero_h(float* __restrict__ hf, bf16* __restrict__ hb, int n) {
  int i = blockIdx.x * blockDim.x + threadIdx.x;
  if (i < n) { hf[i] = 0.0f; hb[i] = (bf16)0.0f; }
}

// encoder step input: out[b,:] = bf16(embedding[x[b,t],:])
__global__ void gather_embed(const float* __restrict__ emb,
                             const int* __restrict__ x, int t,
                             bf16* __restrict__ out) {
  const int b = blockIdx.x, e = threadIdx.x;
  const int tok = x[b * SQ + t];
  out[(size_t)b * EQ + e] = (bf16)emb[(size_t)tok * EQ + e];
}

__global__ void bcast_e0(const float* __restrict__ emb,
                         bf16* __restrict__ out) {   // START token = 0
  const int b = blockIdx.x, e = threadIdx.x;
  out[(size_t)b * EQ + e] = (bf16)emb[e];
}

// GRU gate fusion: gi,gh f32 [B,3H]; hf master f32; hb bf16 for next GEMM.
__global__ void gru_gate(const float* __restrict__ gi,
                         const float* __restrict__ gh,
                         float* __restrict__ hf, bf16* __restrict__ hb) {
  const int i = blockIdx.x * blockDim.x + threadIdx.x;   // B*H threads
  const int b = i >> 10;           // /H
  const int j = i & (HQ - 1);
  const float* gib = gi + (size_t)b * 3 * HQ;
  const float* ghb = gh + (size_t)b * 3 * HQ;
  const float ir = gib[j],           hr = ghb[j];
  const float iz = gib[HQ + j],      hz = ghb[HQ + j];
  const float in_ = gib[2 * HQ + j], hn = ghb[2 * HQ + j];
  const float r = 1.0f / (1.0f + __expf(-(ir + hr)));
  const float z = 1.0f / (1.0f + __expf(-(iz + hz)));
  const float n = tanhf(in_ + r * hn);
  const float h = hf[i];
  const float hnew = (1.0f - z) * n + z * h;
  hf[i] = hnew;
  hb[i] = (bf16)hnew;
}

__global__ void reparam(const float* __restrict__ mu,
                        const float* __restrict__ lv,
                        const float* __restrict__ eps,
                        bf16* __restrict__ z) {
  const int i = blockIdx.x * blockDim.x + threadIdx.x;   // B*L
  z[i] = (bf16)(mu[i] + eps[i] * __expf(0.5f * lv[i]));
}

__global__ void tanh_h(const float* __restrict__ in, float* __restrict__ hf,
                       bf16* __restrict__ hb) {
  const int i = blockIdx.x * blockDim.x + threadIdx.x;   // B*H
  const float v = tanhf(in[i]);
  hf[i] = v;
  hb[i] = (bf16)v;
}

// per-row argmax over V=256 logits (first-max tie-break, like jnp.argmax),
// fused with the embedding gather for the next decoder input.
__global__ void argmax_gather(const float* __restrict__ xrecon, int t,
                              const float* __restrict__ emb,
                              bf16* __restrict__ e_out) {
  __shared__ float sval[VQ];
  __shared__ int   sidx[VQ];
  const int b = blockIdx.x, v = threadIdx.x;
  const float* row = xrecon + (size_t)b * MAXL * VQ + (size_t)t * VQ;
  sval[v] = row[v];
  sidx[v] = v;
  __syncthreads();
  for (int s = VQ / 2; s > 0; s >>= 1) {
    if (v < s) {
      const float v2 = sval[v + s];
      const int   i2 = sidx[v + s];
      if (v2 > sval[v] || (v2 == sval[v] && i2 < sidx[v])) {
        sval[v] = v2; sidx[v] = i2;
      }
    }
    __syncthreads();
  }
  const int tok = sidx[0];
  e_out[(size_t)b * EQ + v] = (bf16)emb[(size_t)tok * EQ + v];
}

// ------------------------------ host side ----------------------------------
static inline size_t al256(size_t x) { return (x + 255) & ~(size_t)255; }

extern "C" void kernel_launch(void* const* d_in, const int* in_sizes, int n_in,
                              void* d_out, int out_size, void* d_ws,
                              size_t ws_size, hipStream_t stream) {
  (void)in_sizes; (void)n_in; (void)out_size; (void)ws_size;
  const int*   x       = (const int*)  d_in[0];
  const float* eps     = (const float*)d_in[1];
  const float* emb     = (const float*)d_in[2];
  const float* encWih  = (const float*)d_in[3];
  const float* encWhh  = (const float*)d_in[4];
  const float* enc_bih = (const float*)d_in[5];
  const float* enc_bhh = (const float*)d_in[6];
  const float* muW     = (const float*)d_in[7];
  const float* mu_b    = (const float*)d_in[8];
  const float* lvW     = (const float*)d_in[9];
  const float* lv_b    = (const float*)d_in[10];
  const float* l2hW    = (const float*)d_in[11];
  const float* l2h_b   = (const float*)d_in[12];
  const float* decWih  = (const float*)d_in[13];
  const float* decWhh  = (const float*)d_in[14];
  const float* dec_bih = (const float*)d_in[15];
  const float* dec_bhh = (const float*)d_in[16];
  const float* outW    = (const float*)d_in[17];
  const float* out_b   = (const float*)d_in[18];

  float* out = (float*)d_out;
  float* out_xrecon = out;                                   // [B,MAXL,V]
  float* out_mu     = out + (size_t)BQ * MAXL * VQ;          // [B,L]
  float* out_lv     = out_mu + (size_t)BQ * LQ;              // [B,L]

  // --- workspace carve-up ---
  char* w = (char*)d_ws;
  size_t off = 0;
  auto carve = [&](size_t bytes) { char* p = w + off; off += al256(bytes); return p; };
  bf16* encWih_bf = (bf16*)carve((size_t)3 * HQ * EQ * 2);
  bf16* encWhh_bf = (bf16*)carve((size_t)3 * HQ * HQ * 2);
  bf16* decWih_bf = (bf16*)carve((size_t)3 * HQ * EQ * 2);
  bf16* decWhh_bf = (bf16*)carve((size_t)3 * HQ * HQ * 2);
  bf16* outW_bf   = (bf16*)carve((size_t)VQ * HQ * 2);
  bf16* muW_bf    = (bf16*)carve((size_t)LQ * HQ * 2);
  bf16* lvW_bf    = (bf16*)carve((size_t)LQ * HQ * 2);
  bf16* l2hW_bf   = (bf16*)carve((size_t)HQ * LQ * 2);
  float* hf       = (float*)carve((size_t)BQ * HQ * 4);
  bf16*  hb       = (bf16*) carve((size_t)BQ * HQ * 2);
  bf16*  xt_bf    = (bf16*) carve((size_t)BQ * EQ * 2);
  float* gi       = (float*)carve((size_t)BQ * 3 * HQ * 4);
  float* gh       = (float*)carve((size_t)BQ * 3 * HQ * 4);
  bf16*  z_bf     = (bf16*) carve((size_t)BQ * LQ * 2);
  float* tmpH     = (float*)carve((size_t)BQ * HQ * 4);
  bf16*  e_bf     = (bf16*) carve((size_t)BQ * EQ * 2);

  const dim3 blk256(256);
  auto castN = [&](const float* src, bf16* dst, int n) {
    cast_f32_bf16<<<dim3((n + 255) / 256), blk256, 0, stream>>>(src, dst, n);
  };

  // --- 0. weight casts (once per call) ---
  castN(encWih, encWih_bf, 3 * HQ * EQ);
  castN(encWhh, encWhh_bf, 3 * HQ * HQ);
  castN(decWih, decWih_bf, 3 * HQ * EQ);
  castN(decWhh, decWhh_bf, 3 * HQ * HQ);
  castN(outW,   outW_bf,   VQ * HQ);
  castN(muW,    muW_bf,    LQ * HQ);
  castN(lvW,    lvW_bf,    LQ * HQ);
  castN(l2hW,   l2hW_bf,   HQ * LQ);
  zero_h<<<dim3((BQ * HQ) / 256), blk256, 0, stream>>>(hf, hb, BQ * HQ);

  const dim3 gGate((BQ * HQ) / 256);
  const dim3 gGemm3H(3 * HQ / TN, BQ / TM);   // (12,16)  N=3072
  const dim3 gGemmL(LQ / TN, BQ / TM);        // (1,16)   N=256
  const dim3 gGemmH(HQ / TN, BQ / TM);        // (4,16)   N=1024

  // --- 1. encoder scan ---
  for (int t = 0; t < SQ; ++t) {
    gather_embed<<<dim3(BQ), dim3(EQ), 0, stream>>>(emb, x, t, xt_bf);
    gemm_bf16_wmma<<<gGemm3H, blk256, 0, stream>>>(
        xt_bf, EQ, encWih_bf, EQ, enc_bih, gi, 3 * HQ, EQ);
    gemm_bf16_wmma<<<gGemm3H, blk256, 0, stream>>>(
        hb, HQ, encWhh_bf, HQ, enc_bhh, gh, 3 * HQ, HQ);
    gru_gate<<<gGate, blk256, 0, stream>>>(gi, gh, hf, hb);
  }

  // --- 2. latent head: mu / logvar straight into d_out, then reparam ---
  gemm_bf16_wmma<<<gGemmL, blk256, 0, stream>>>(
      hb, HQ, muW_bf, HQ, mu_b, out_mu, LQ, HQ);
  gemm_bf16_wmma<<<gGemmL, blk256, 0, stream>>>(
      hb, HQ, lvW_bf, HQ, lv_b, out_lv, LQ, HQ);
  reparam<<<dim3((BQ * LQ) / 256), blk256, 0, stream>>>(out_mu, out_lv, eps, z_bf);

  // --- 3. decoder init: h = tanh(z @ l2h^T + b), e0 = embedding[START] ---
  gemm_bf16_wmma<<<gGemmH, blk256, 0, stream>>>(
      z_bf, LQ, l2hW_bf, LQ, l2h_b, tmpH, HQ, LQ);
  tanh_h<<<gGate, blk256, 0, stream>>>(tmpH, hf, hb);
  bcast_e0<<<dim3(BQ), dim3(EQ), 0, stream>>>(emb, e_bf);

  // --- 4. greedy autoregressive decode ---
  for (int t = 0; t < MAXL; ++t) {
    gemm_bf16_wmma<<<gGemm3H, blk256, 0, stream>>>(
        e_bf, EQ, decWih_bf, EQ, dec_bih, gi, 3 * HQ, EQ);
    gemm_bf16_wmma<<<gGemm3H, blk256, 0, stream>>>(
        hb, HQ, decWhh_bf, HQ, dec_bhh, gh, 3 * HQ, HQ);
    gru_gate<<<gGate, blk256, 0, stream>>>(gi, gh, hf, hb);
    // logits written directly into x_recon[b, t, :] (row stride MAXL*V)
    gemm_bf16_wmma<<<gGemmL, blk256, 0, stream>>>(
        hb, HQ, outW_bf, HQ, out_b, out_xrecon + (size_t)t * VQ, MAXL * VQ, HQ);
    argmax_gather<<<dim3(BQ), dim3(VQ), 0, stream>>>(out_xrecon, t, emb, e_bf);
  }
}
